// SimpleMultiTaskDecoders_40561671143598
// MI455X (gfx1250) — matrix operand order repaired and checked by reading, not compiled, not run on hardware
//
#include <hip/hip_runtime.h>

// ---------------------------------------------------------------------------
// Fused 4-expert 3-layer MLP decoder for MI455X (gfx1250, wave32, WMMA).
//   pre-pass: W1/W2 fp32 -> bf16, pre-transposed into d_ws (1.6 MB, L2-resident)
//   L1: H1 = relu(x[64t,2048] @ W1cat[2048,384] + b1cat)      (bf16 WMMA)
//   L2: H2 = relu(block-diag expert GEMMs, K=64/128 -> 192)   (bf16 WMMA)
//   L3: out = <H2_e, W3_e> + b3_e  selected per-row by label  (scalar)
// All intermediates live in LDS; only x (512MB) streams from HBM.
// Weight tiles are staged global->LDS with GLOBAL_LOAD_ASYNC_TO_LDS_B128
// (ASYNCcnt) when the toolchain exposes the builtin.
// ---------------------------------------------------------------------------

typedef __attribute__((ext_vector_type(16))) __bf16 v16bf;
typedef __attribute__((ext_vector_type(8)))  float  v8f;
typedef __attribute__((ext_vector_type(4)))  int    v4i;

#define INPUT_DIM 2048
#define HTOT      384     // fused layer-1 width (64+64+128+128)
#define H2TOT     192     // fused layer-2 width (32+32+64+64)
#define BM        64      // rows per block
#define BK        32      // K chunk (WMMA bf16 K)
#define THREADS   256     // 8 waves

#define W1T_ELEMS (HTOT * INPUT_DIM)   // 786432 bf16
#define W2T_ELEMS 20480                // bf16

// dynamic LDS layout (bytes):
//  [0,      49152)  H1 tile : 64 x 384 bf16
//  [49152,  90112)  W region: phase1 W1 tile (col-major 384x32),
//                            phase2 W2^T (20480 bf16 = 40960B)
//  [90112, 114688)  C region: phase1 x tile (64x32 bf16),
//                            phase2/3 H2 tile (64x192 bf16)
#define LDS_H1_OFF 0
#define LDS_W_OFF  49152
#define LDS_C_OFF  90112
#define LDS_BYTES  114688

#if defined(__gfx1250__) && __has_builtin(__builtin_amdgcn_global_load_async_to_lds_b128)
#define HAVE_ASYNC_LDS 1
#else
#define HAVE_ASYNC_LDS 0
#endif

#define AS_GLOBAL __attribute__((address_space(1)))
#define AS_SHARED __attribute__((address_space(3)))

struct Params {
  const float* x;
  const int*   labels;
  const float* W1[4];
  const float* b1[4];
  const float* W2[4];
  const float* b2[4];
  const float* W3[4];
  const float* b3[4];
  float* out;
  int n;
};

union FragU { v16bf v; uint4 q[2]; };
union Pack8 { unsigned short s[8]; uint4 q; };
union Pack4 { unsigned short s[4]; uint2 q; };

__device__ __forceinline__ unsigned short f2bf(float f) {
  unsigned int u = __builtin_bit_cast(unsigned int, f);
  unsigned int r = u + 0x7FFFu + ((u >> 16) & 1u);   // round-to-nearest-even
  return (unsigned short)(r >> 16);
}
__device__ __forceinline__ float bf2f(unsigned short h) {
  unsigned int u = ((unsigned int)h) << 16;
  return __builtin_bit_cast(float, u);
}

// copy 16B global -> LDS; async (ASYNCcnt) when available
__device__ __forceinline__ void cp16(unsigned short* ldsDst, const unsigned short* gSrc) {
#if HAVE_ASYNC_LDS
  v4i* gs = (v4i*)(void*)(size_t)(const void*)gSrc;   // strip const, retype
  v4i* ls = (v4i*)(void*)ldsDst;
  __builtin_amdgcn_global_load_async_to_lds_b128(
      (AS_GLOBAL v4i*)gs, (AS_SHARED v4i*)ls, 0, 0);
#else
  *(uint4*)ldsDst = *(const uint4*)gSrc;
#endif
}
__device__ __forceinline__ void wait_cp() {
#if HAVE_ASYNC_LDS
#if __has_builtin(__builtin_amdgcn_s_wait_asynccnt)
  __builtin_amdgcn_s_wait_asynccnt(0);
#else
  asm volatile("s_wait_asynccnt 0" ::: "memory");
#endif
#endif
}

// A-matrix frag (16-bit, 16x32): lanes 0-15 -> K kBase+[0..7]&[16..23],
// lanes 16-31 -> K kBase+[8..15]&[24..31]; row = M (per ISA 7.12.2).
__device__ __forceinline__ v16bf fragA(const unsigned short* p, int row,
                                       int stride, int kBase, int lane) {
  const unsigned short* q = p + row * stride + kBase + ((lane & 16) ? 8 : 0);
  FragU u;
  u.q[0] = *(const uint4*)(q);
  u.q[1] = *(const uint4*)(q + 16);
  return u.v;
}

// B-matrix frag (16-bit, 32x16) from a K-contiguous (transposed) LDS image:
// lane n%16 -> column N, lanes 0-15 K kBase+[0..7]&[16..23], lanes 16-31 K kBase+[8..15]&[24..31].
__device__ __forceinline__ v16bf fragB(const unsigned short* colMajor, int stride,
                                       int kBase, int lane) {
  const unsigned short* q = colMajor + (lane & 15) * stride + kBase + ((lane & 16) ? 8 : 0);
  FragU u;
  u.q[0] = *(const uint4*)(q);
  u.q[1] = *(const uint4*)(q + 16);
  return u.v;
}

__device__ __forceinline__ int expOf(int col) {          // fused L1 col -> expert
  return (col < 64) ? 0 : (col < 128) ? 1 : (col < 256) ? 2 : 3;
}

// ---------------------------------------------------------------------------
// Pre-pass: build bf16, K-contiguous weight images in workspace.
//   W1t[col][k]  col 0..383, k 0..2047              (ws elems [0, 786432))
//   W2t: per-expert [col][k] blocks at {0,2048,4096,12288}  (ws + 786432)
// ---------------------------------------------------------------------------
__global__ __launch_bounds__(THREADS)
void convert_weights_kernel(Params P, unsigned short* W1t, unsigned short* W2t) {
  int idx = blockIdx.x * THREADS + threadIdx.x;
  if (idx < HTOT * (INPUT_DIM / 4)) {          // 196608: one col x 4 k's each
    int col = idx >> 9;                        // 0..383
    int kq  = (idx & 511) << 2;                // 0..2044
    int e   = expOf(col);
    int h   = (e < 2) ? 64 : 128;
    int off = (e < 2) ? e * 64 : (e - 1) * 128;
    const float* w = P.W1[e] + (col - off);
    Pack4 p;
    p.s[0] = f2bf(w[(size_t)(kq + 0) * h]);
    p.s[1] = f2bf(w[(size_t)(kq + 1) * h]);
    p.s[2] = f2bf(w[(size_t)(kq + 2) * h]);
    p.s[3] = f2bf(w[(size_t)(kq + 3) * h]);
    *(uint2*)(W1t + (size_t)col * INPUT_DIM + kq) = p.q;
  } else if (idx < HTOT * (INPUT_DIM / 4) + W2T_ELEMS / 4) {
    int u = idx - HTOT * (INPUT_DIM / 4);      // 0..5119
    int e, local;
    if (u < 512)       { e = 0; local = u; }
    else if (u < 1024) { e = 1; local = u - 512; }
    else if (u < 3072) { e = 2; local = u - 1024; }
    else               { e = 3; local = u - 3072; }
    int h    = (e < 2) ? 64 : 128;
    int h2   = h >> 1;
    int sh   = (e < 2) ? 4 : 5;                // k-quads per col: 16 or 32
    int col  = local >> sh;
    int k    = (local & ((1 << sh) - 1)) << 2;
    int base = (e < 2) ? e * 2048 : 4096 + (e - 2) * 8192;
    const float* w = P.W2[e] + col;
    Pack4 p;
    p.s[0] = f2bf(w[(k + 0) * h2]);
    p.s[1] = f2bf(w[(k + 1) * h2]);
    p.s[2] = f2bf(w[(k + 2) * h2]);
    p.s[3] = f2bf(w[(k + 3) * h2]);
    *(uint2*)(W2t + base + col * h + k) = p.q;
  }
}

// ---------------------------------------------------------------------------
// Main fused kernel.
// ---------------------------------------------------------------------------
__global__ __launch_bounds__(THREADS)
void moe_decoders_kernel(Params P, const unsigned short* W1t, const unsigned short* W2t) {
  extern __shared__ char smem[];
  unsigned short* ldsH1 = (unsigned short*)(smem + LDS_H1_OFF);
  unsigned short* ldsW  = (unsigned short*)(smem + LDS_W_OFF);
  unsigned short* ldsX  = (unsigned short*)(smem + LDS_C_OFF);
  unsigned short* ldsH2 = (unsigned short*)(smem + LDS_C_OFF);

  const int tid     = threadIdx.x;
  const int lane    = tid & 31;
  const int lane15  = lane & 15;
  const int wave    = tid >> 5;         // 0..7
  const int strip   = wave >> 1;        // 0..3 : 16-row strip
  const int colHalf = wave & 1;         // 0..1 : 192-col half
  const int row0    = blockIdx.x * BM;

  // ---------------- Phase 1: H1 = relu(x @ W1cat + b1cat) -----------------
  v8f acc[12];
#pragma unroll
  for (int i = 0; i < 12; ++i)
    acc[i] = (v8f){0.f, 0.f, 0.f, 0.f, 0.f, 0.f, 0.f, 0.f};

  for (int kc = 0; kc < INPUT_DIM; kc += BK) {
    // stage x tile 64x32 (fp32 -> bf16), fully coalesced b128 traffic
    {
      int r  = tid >> 2;
      int kb = (tid & 3) * 8;
      const float* src = P.x + (size_t)(row0 + r) * INPUT_DIM + kc + kb;
      float4 v0 = *(const float4*)(src);
      float4 v1 = *(const float4*)(src + 4);
      Pack8 p8;
      p8.s[0] = f2bf(v0.x); p8.s[1] = f2bf(v0.y);
      p8.s[2] = f2bf(v0.z); p8.s[3] = f2bf(v0.w);
      p8.s[4] = f2bf(v1.x); p8.s[5] = f2bf(v1.y);
      p8.s[6] = f2bf(v1.z); p8.s[7] = f2bf(v1.w);
      *(uint4*)(ldsX + r * BK + kb) = p8.q;
      if (kc + BK < INPUT_DIM)
        __builtin_prefetch(src + BK, 0, 1);          // global_prefetch_b8
    }
    // stage W1t K-slab: ldsW[col*32 + k], pure 16B copies (already bf16)
#pragma unroll
    for (int j = 0; j < 6; ++j) {
      int u    = tid + j * THREADS;                  // 0..1535 uint4 units
      int col  = u >> 2;
      int part = (u & 3) * 8;
      cp16(ldsW + col * BK + part,
           W1t + (size_t)col * INPUT_DIM + kc + part);
    }
    wait_cp();
    __syncthreads();

    v16bf a = fragA(ldsX, strip * 16 + lane15, BK, 0, lane);
#pragma unroll
    for (int ct = 0; ct < 12; ++ct) {
      int colBase = (colHalf * 12 + ct) * 16;
      v16bf b = fragB(ldsW + colBase * BK, BK, 0, lane);
      acc[ct] = __builtin_amdgcn_wmma_f32_16x16x32_bf16(
          false, a, false, b, (short)0, acc[ct], false, false);
    }
    __syncthreads();
  }

  // epilogue: +b1, relu, bf16 -> ldsH1  (C layout: vgpr i -> M=i / M=8+i)
#pragma unroll
  for (int ct = 0; ct < 12; ++ct) {
    int col = (colHalf * 12 + ct) * 16 + lane15;
    int e   = expOf(col);
    int off = (e < 2) ? e * 64 : (e - 1) * 128;
    float bias = P.b1[e][col - off];
    int rbase  = strip * 16 + ((lane & 16) ? 8 : 0);
#pragma unroll
    for (int i = 0; i < 8; ++i) {
      float v = acc[ct][i] + bias;
      v = v > 0.f ? v : 0.f;
      ldsH1[(rbase + i) * HTOT + col] = f2bf(v);
    }
  }

  // stage W2^T into ldsW (same layout as workspace image: linear copy)
#pragma unroll
  for (int j = 0; j < 10; ++j) {
    int u = tid + j * THREADS;                       // 0..2559 uint4 units
    cp16(ldsW + u * 8, W2t + u * 8);
  }
  wait_cp();
  __syncthreads();

  // ---------------- Phase 2: H2 = relu(H1_e @ W2_e + b2_e) ----------------
  // 48 output tiles (sc:8, st:8, wm:16, ch:16), 6 per wave.
#pragma unroll
  for (int i = 0; i < 6; ++i) {
    int t = wave * 6 + i;
    int e, s2, ct;
    if (t < 8)       { e = 0; s2 = t >> 1;        ct = t & 1; }
    else if (t < 16) { e = 1; s2 = (t - 8) >> 1;  ct = (t - 8) & 1; }
    else if (t < 32) { e = 2; s2 = (t - 16) >> 2; ct = (t - 16) & 3; }
    else             { e = 3; s2 = (t - 32) >> 2; ct = (t - 32) & 3; }
    int h    = (e < 2) ? 64 : 128;
    int off1 = (e < 2) ? e * 64 : (e - 1) * 128;
    int off2 = off1 >> 1;
    int wb   = (e < 2) ? e * 2048 : 4096 + (e - 2) * 8192;

    v8f acc2 = (v8f){0.f, 0.f, 0.f, 0.f, 0.f, 0.f, 0.f, 0.f};
    for (int kk = 0; kk < h; kk += 32) {
      v16bf a = fragA(ldsH1, s2 * 16 + lane15, HTOT, off1 + kk, lane);
      v16bf b = fragB(ldsW + wb + (ct * 16) * h, h, kk, lane);
      acc2 = __builtin_amdgcn_wmma_f32_16x16x32_bf16(
          false, a, false, b, (short)0, acc2, false, false);
    }
    int col    = ct * 16 + lane15;
    float bias = P.b2[e][col];
    int rbase  = s2 * 16 + ((lane & 16) ? 8 : 0);
#pragma unroll
    for (int r = 0; r < 8; ++r) {
      float v = acc2[r] + bias;
      v = v > 0.f ? v : 0.f;
      ldsH2[(rbase + r) * H2TOT + off2 + col] = f2bf(v);
    }
  }
  __syncthreads();

  // ---------------- Phase 3: per-row W3 dot + label select ----------------
  if (tid < BM) {
    int rowg = row0 + tid;
    int lab  = P.labels[rowg];
    float res = 0.f;
    if (lab >= 0 && lab < 4) {
      int h2   = (lab < 2) ? 32 : 64;
      int off2 = (lab < 2) ? lab * 32 : (lab - 1) * 64;
      const float* w3 = P.W3[lab];
      float s = 0.f;
      for (int c = 0; c < h2; ++c)
        s += bf2f(ldsH2[tid * H2TOT + off2 + c]) * w3[c];
      res = s + P.b3[lab][0];
    }
    P.out[rowg] = res;
  }
}

extern "C" void kernel_launch(void* const* d_in, const int* in_sizes, int n_in,
                              void* d_out, int out_size, void* d_ws, size_t ws_size,
                              hipStream_t stream) {
  Params P;
  P.x      = (const float*)d_in[0];
  P.labels = (const int*)d_in[1];
  for (int e = 0; e < 4; ++e) {
    P.W1[e] = (const float*)d_in[2 + e * 6 + 0];
    P.b1[e] = (const float*)d_in[2 + e * 6 + 1];
    P.W2[e] = (const float*)d_in[2 + e * 6 + 2];
    P.b2[e] = (const float*)d_in[2 + e * 6 + 3];
    P.W3[e] = (const float*)d_in[2 + e * 6 + 4];
    P.b3[e] = (const float*)d_in[2 + e * 6 + 5];
  }
  P.out = (float*)d_out;
  P.n   = in_sizes[1];

  unsigned short* W1t = (unsigned short*)d_ws;
  unsigned short* W2t = W1t + W1T_ELEMS;

  // pre-pass: weights -> bf16, K-contiguous (196608 + 5120 work items)
  int preItems  = HTOT * (INPUT_DIM / 4) + W2T_ELEMS / 4;
  int preBlocks = (preItems + THREADS - 1) / THREADS;
  convert_weights_kernel<<<preBlocks, THREADS, 0, stream>>>(P, W1t, W2t);

  int blocks = (P.n + BM - 1) / BM;
  moe_decoders_kernel<<<blocks, THREADS, LDS_BYTES, stream>>>(P, W1t, W2t);
}
